// MetricLayer_66675072303286
// MI455X (gfx1250) — compile-verified, matrix operand order checked by reading, not run on hardware
//
#include <hip/hip_runtime.h>
#include <hip/hip_bf16.h>
#include <math.h>

// ---- problem constants (from reference) ----
#define ROWLEN 1000      // 1 positive + 999 negatives per user
#define NUSERS 8192
#define TOPK   10
#define BIGNEG (-3.4028234663852886e38f)   // jnp.finfo(float32).min == -FLT_MAX

typedef __attribute__((ext_vector_type(8))) int v8i;

// One wave32 per user. Each lane streams 16 float4 (2 logit pairs) + 16 int2
// (2 dup flags), accumulates:
//   cnt  = #elements with masked_logit >  masked_logit[0]   (== position)
//   dsum = #dup elements
// Wave reduction of both counters runs on the matrix pipe:
//   V_WMMA_I32_16X16X64_IU8 with B = ones sums all 1024 A-bytes into row sums
//   replicated across columns; per-lane sum of the 8 D VGPRs + one xor-16
//   shuffle gives the wave-wide total in every lane.
__global__ __launch_bounds__(256) void metric_kernel(
    const float* __restrict__ logits,   // (NUSERS*ROWLEN, 1, 2) f32
    const int*   __restrict__ dup,      // (NUSERS*ROWLEN, 1, 1) i32
    float*       __restrict__ out)      // [in_top_k | ndcg | weights], each NUSERS
{
  const int wave = threadIdx.x >> 5;
  const int lane = threadIdx.x & 31;
  const int user = (int)blockIdx.x * 8 + wave;       // grid = NUSERS/8 exactly
  const long base = (long)user * ROWLEN;             // element index of item 0

  // Threshold = masked logit of the positive item (uniform address, L2/L0 hit).
  const int   d0 = dup[base];
  const float l0 = logits[base * 2 + 1];
  const float th = d0 ? BIGNEG : l0;

  // user base is a multiple of 2 elements -> float4/int2 aligned (8000B/4000B).
  const float4* lp = (const float4*)(logits) + (base >> 1);
  const int2*   dp = (const int2*)(dup)     + (base >> 1);

  int cnt = 0, dsum = 0;
#pragma unroll 4
  for (int p = lane; p < ROWLEN / 2; p += 32) {      // 500 pairs per user
    float4 v = lp[p];                                 // b128, 512B/wave coalesced
    int2   m = dp[p];                                 // b64,  256B/wave coalesced
    float a = m.x ? BIGNEG : v.y;                     // channel-1 logits
    float b = m.y ? BIGNEG : v.w;
    cnt  += (a > th) ? 1 : 0;                         // strict > == stable-sort rank
    cnt  += (b > th) ? 1 : 0;
    dsum += m.x + m.y;
  }
  // cnt, dsum <= 32 per lane -> fit u8 exactly.

  // ---- wave-wide integer reduction via V_WMMA_I32_16X16X64_IU8 ----
  v8i B = {0x01010101, 0x01010101, 0x01010101, 0x01010101,
           0x01010101, 0x01010101, 0x01010101, 0x01010101}; // all-ones B matrix
  v8i A1 = {}; A1[0] = cnt;    // one count byte per lane somewhere in A
  v8i A2 = {}; A2[0] = dsum;
  v8i Z  = {};
  // 7 args: (sgn_a, A, sgn_b, B, C, reuse_a, reuse_b)
  v8i C1 = __builtin_amdgcn_wmma_i32_16x16x64_iu8(false, A1, false, B, Z, false, false);
  v8i C2 = __builtin_amdgcn_wmma_i32_16x16x64_iu8(false, A2, false, B, Z, false, false);

  int r1 = C1[0] + C1[1] + C1[2] + C1[3] + C1[4] + C1[5] + C1[6] + C1[7];
  int r2 = C2[0] + C2[1] + C2[2] + C2[3] + C2[4] + C2[5] + C2[6] + C2[7];
  r1 += __shfl_xor(r1, 16, 32);   // combine rows 0-7 with rows 8-15
  r2 += __shfl_xor(r2, 16, 32);

  if (lane == 0) {
    float pos = (float)r1;                               // position of item 0
    float itk = (r1 < TOPK) ? 1.0f : 0.0f;
    float nd  = itk * (0.6931471805599453f / logf(pos + 2.0f)); // ln2/ln(p+2)
    out[user]              = itk;
    out[NUSERS + user]     = nd;
    out[2 * NUSERS + user] = (r2 != ROWLEN - 1) ? 1.0f : 0.0f;  // Σdup != 999
  }
}

extern "C" void kernel_launch(void* const* d_in, const int* in_sizes, int n_in,
                              void* d_out, int out_size, void* d_ws, size_t ws_size,
                              hipStream_t stream) {
  (void)in_sizes; (void)n_in; (void)out_size; (void)d_ws; (void)ws_size;
  const float* logits = (const float*)d_in[0];
  const int*   dup    = (const int*)d_in[1];
  float*       out    = (float*)d_out;
  // 1024 blocks x 8 waves = 8192 waves, one per user; all outputs fully written.
  hipLaunchKernelGGL(metric_kernel, dim3(NUSERS / 8), dim3(256), 0, stream,
                     logits, dup, out);
}
